// ViSNetDynamics_12979391168635
// MI455X (gfx1250) — matrix operand order repaired and checked by reading, not compiled, not run on hardware
//
#include <hip/hip_runtime.h>
#include <hip/hip_bf16.h>
#include <math.h>

typedef _Float16 h16;
typedef __attribute__((ext_vector_type(16))) _Float16 v16h;
typedef __attribute__((ext_vector_type(8)))  _Float16 v8h;
typedef __attribute__((ext_vector_type(8)))  float     v8f;

#define NA_C 512
#define NR_C 1536
#define NN_C 2048
#define E_C  24576

__device__ __forceinline__ float siluf(float x){ return x / (1.0f + expf(-x)); }
__device__ __forceinline__ float sigmf(float x){ return 1.0f / (1.0f + expf(-x)); }

// ---------------- generic WMMA GEMM ----------------
enum { EPI_PLAIN=0, EPI_SILU_F16=1, EPI_F16=2, EPI_NBR=3, EPI_EDGEEMB=4, EPI_FUPD=5 };

struct GemmP {
  const h16* A; long lda;
  const h16* WT; int K;
  const float* bias;
  float* outF; long ldoF;
  h16*  outH; long ldoH;
  int M, n0base, epi;
  const int* ei; int E;
  const float* auxC;   // NBR: C*em
  const float* auxX;   // NBR / EDGEEMB: x (N x 256)
  float* auxAcc;       // NBR: msg accumulator (N x 256)
  const float* fijOld; // FUPD
  const float* em;     // FUPD
  const h16* wtrg;     // FUPD (8N x 256), used at src
  const h16* wsrc;     // FUPD (8N x 256), used at dst
};

__device__ __forceinline__ v16h ldfrag(const h16* p0, const h16* p1){
  v8h lo = *(const v8h*)p0;
  v8h hh = *(const v8h*)p1;
  return __builtin_shufflevector(lo, hh, 0,1,2,3,4,5,6,7,8,9,10,11,12,13,14,15);
}

template<int NT>
__global__ __launch_bounds__(128) void k_gemm(GemmP p){
  const int wave = threadIdx.x >> 5;
  const int lane = threadIdx.x & 31;
  const int row0 = (blockIdx.x * 4 + wave) * 16;
  if (row0 >= p.M) return;
  const int n0 = p.n0base + blockIdx.y * (NT * 16);
  const int ll = lane & 15;
  const int hi = lane >> 4;               // half-wave select
  const h16* Ab = p.A + (long)(row0 + ll) * p.lda + hi * 8;
  const h16* Bb0 = p.WT + (long)(n0 + ll) * p.K + hi * 8;
  v8f acc[NT];
  #pragma unroll
  for (int t = 0; t < NT; t++) acc[t] = v8f{};
  for (int k0 = 0; k0 < p.K; k0 += 32){
    v16h a = ldfrag(Ab + k0, Ab + k0 + 16);
    #pragma unroll
    for (int t = 0; t < NT; t++){
      const h16* Bb = Bb0 + (long)(t*16) * p.K + k0;
      v16h b = ldfrag(Bb, Bb + 16);
      acc[t] = __builtin_amdgcn_wmma_f32_16x16x32_f16(false, a, false, b,
                                                      (short)0, acc[t], false, false);
    }
  }
  #pragma unroll
  for (int t = 0; t < NT; t++){
    const int col = n0 + t*16 + ll;
    const float bv = p.bias ? p.bias[col] : 0.f;
    #pragma unroll
    for (int r = 0; r < 8; r++){
      const int m = row0 + r + hi*8;
      float v = acc[t][r] + bv;
      if (p.epi == EPI_PLAIN){
        p.outF[(long)m * p.ldoF + col] = v;
      } else if (p.epi == EPI_SILU_F16){
        p.outH[(long)m * p.ldoH + col] = (h16)siluf(v);
      } else if (p.epi == EPI_F16){
        p.outH[(long)m * p.ldoH + col] = (h16)v;
      } else if (p.epi == EPI_NBR){
        const int s = p.ei[m], d = p.ei[p.E + m];
        float wd = v * p.auxC[m];
        atomicAdd(p.auxAcc + (long)s*256 + col, p.auxX[(long)d*256 + col] * wd);
      } else if (p.epi == EPI_EDGEEMB){
        const int s = p.ei[m], d = p.ei[p.E + m];
        float f = v * (p.auxX[(long)s*256 + col] + p.auxX[(long)d*256 + col]);
        p.outF[(long)m*256 + col] = f;
        p.outH[(long)m*256 + col] = (h16)f;
      } else { // EPI_FUPD
        const int s = p.ei[m], d = p.ei[p.E + m];
        float wd = 0.f;
        #pragma unroll
        for (int q8 = 0; q8 < 8; q8++)
          wd += (float)p.wtrg[((long)s*8 + q8)*256 + col] *
                (float)p.wsrc[((long)d*8 + q8)*256 + col];
        float nf = p.fijOld[(long)m*256 + col] + siluf(v) * wd * p.em[m];
        p.outF[(long)m*256 + col] = nf;
        p.outH[(long)m*256 + col] = (h16)nf;
      }
    }
  }
}

// weight prep: WT[n][k] = (h16) W[k*ldw + coff + n]
__global__ void k_prep(const float* W, int ldw, int coff, int K, int Nsub, h16* WT){
  long i = (long)blockIdx.x*256 + threadIdx.x;
  if (i >= (long)K*Nsub) return;
  int n = (int)(i / K), k = (int)(i % K);
  WT[(long)n*K + k] = (h16)W[(long)k*ldw + coff + n];
}

__global__ void k_nlin(const float* A, int lda, int aoff, int K, int Nc,
                       const float* W, const float* b, float* out, int M, int act){
  long i = (long)blockIdx.x*256 + threadIdx.x;
  if (i >= (long)M*Nc) return;
  int m = (int)(i / Nc), n = (int)(i % Nc);
  float s = b ? b[n] : 0.f;
  for (int k = 0; k < K; k++) s += A[(long)m*lda + aoff + k] * W[(long)k*Nc + n];
  out[i] = act ? siluf(s) : s;
}

__global__ void k_lnorm(const float* in, const float* g, const float* b,
                        float* outF, h16* outH, int C, int act){
  __shared__ float red[256];
  const int row = blockIdx.x, t = threadIdx.x;
  float x = in[(long)row*C + t];
  red[t] = x; __syncthreads();
  for (int s = C>>1; s > 0; s >>= 1){ if (t < s) red[t] += red[t+s]; __syncthreads(); }
  float mean = red[0] / C; __syncthreads();
  float d = x - mean; red[t] = d*d; __syncthreads();
  for (int s = C>>1; s > 0; s >>= 1){ if (t < s) red[t] += red[t+s]; __syncthreads(); }
  float var = red[0] / C;
  float y = d * rsqrtf(var + 1e-5f) * g[t] + b[t];
  if (act) y = siluf(y);
  if (outF) outF[(long)row*C + t] = y;
  if (outH) outH[(long)row*C + t] = (h16)y;
}

__global__ void k_cvt(const float* in, long ldi, h16* out, long ldo, long coff, int M, int C){
  long i = (long)blockIdx.x*256 + threadIdx.x;
  if (i >= (long)M*C) return;
  int m = (int)(i / C), c = (int)(i % C);
  out[(long)m*ldo + coff + c] = (h16)in[(long)m*ldi + c];
}

__global__ void k_te(const float* t, const float* tw, const float* tb, float* teval){
  __shared__ float red[128];
  int j = threadIdx.x;
  float f = expf((float)j * (-logf(10000.f) / 127.f));
  float a = t[0] * f;
  red[j] = sinf(a)*tw[j] + cosf(a)*tw[128+j];
  __syncthreads();
  for (int s = 64; s > 0; s >>= 1){ if (j < s) red[j] += red[j+s]; __syncthreads(); }
  if (j == 0) teval[0] = red[0] + tb[0];
}

__global__ void k_bias2(const float* inpb, const float* inpw, const float* teval, float* bias2){
  int c = blockIdx.x*256 + threadIdx.x; if (c >= 256) return;
  bias2[c] = inpb[c] + teval[0]*inpw[256*256 + c];
}

__global__ void k_edge(const int* ei, const int* et, const float* em,
                       const float* xa, const float* xr,
                       const float* betas, const float* means, const float* etype,
                       float* Cbuf, float* dij, h16* eattr, int E){
  int e = blockIdx.x*256 + threadIdx.x; if (e >= E) return;
  int s = ei[e], d = ei[E + e];
  float ps[3], pd[3];
  #pragma unroll
  for (int k = 0; k < 3; k++){
    ps[k] = (s < NA_C) ? xa[(long)s*19 + k] : xr[(long)(s-NA_C)*27 + k];
    pd[k] = (d < NA_C) ? xa[(long)d*19 + k] : xr[(long)(d-NA_C)*27 + k];
  }
  float vx = ps[0]-pd[0], vy = ps[1]-pd[1], vz = ps[2]-pd[2];
  float w = sqrtf(vx*vx + vy*vy + vz*vz + 1e-16f);
  float cut = 0.5f*(cosf(3.14159265358979f*w/8.f) + 1.f) * (w < 8.f ? 1.f : 0.f);
  float m = em[e];
  Cbuf[e] = cut * m;
  float iw = 1.f/w;
  float ux = vx*iw, uy = vy*iw, uz = vz*iw;
  const float s3 = 1.73205080757f;
  float sh[8] = {ux, uy, uz, s3*ux*uz, s3*ux*uy, uy*uy - 0.5f*(ux*ux + uz*uz),
                 s3*uy*uz, 0.86602540378f*(uz*uz - ux*ux)};
  #pragma unroll
  for (int k = 0; k < 8; k++) dij[(long)e*8 + k] = sh[k]*m;
  float ew = expf(-0.625f*w);
  int ty = et[e];
  for (int k = 0; k < 32; k++){
    float dd = ew - means[k];
    float r = cut * expf(-betas[k]*dd*dd);
    eattr[(long)e*32 + k] = (h16)((r + etype[ty*32 + k]) * m);
  }
}

__global__ void k_attn(const float* q, const float* kk, const h16* dk, const float* Cbuf,
                       const int* ei, float* attn, int E){
  long i = (long)blockIdx.x*256 + threadIdx.x; if (i >= (long)E*8) return;
  int e = (int)(i >> 3), hh = (int)(i & 7);
  int s = ei[e], d = ei[E + e];
  const float* qp = q + (long)s*256 + hh*32;
  const float* kp = kk + (long)d*256 + hh*32;
  const h16* dkp = dk + (long)e*256 + hh*32;
  float acc = 0.f;
  for (int j = 0; j < 32; j++) acc += qp[j]*kp[j]*(float)dkp[j];
  attn[i] = siluf(acc) * Cbuf[e];
}

__global__ void k_vmsg(const float* v, const h16* dv, const float* attn, const int* ei,
                       h16* vmsgH, float* xagg, int E){
  long i = (long)blockIdx.x*256 + threadIdx.x; if (i >= (long)E*256) return;
  int e = (int)(i >> 8), c = (int)(i & 255);
  int s = ei[e], d = ei[E + e];
  float m = v[(long)d*256 + c] * (float)dv[i] * attn[(long)e*8 + (c >> 5)];
  vmsgH[i] = (h16)m;
  atomicAdd(xagg + (long)s*256 + c, m);
}

__global__ void k_vscat(const float* vec, const h16* sbuf, const float* dij, const float* em,
                        const int* ei, float* vecagg, int E){
  long i = (long)blockIdx.x*256 + threadIdx.x; if (i >= (long)E*256) return;
  int e = (int)(i >> 8), c = (int)(i & 255);
  if (em[e] == 0.f) return;
  int s = ei[e], d = ei[E + e];
  float s1 = (float)sbuf[(long)e*512 + c], s2 = (float)sbuf[(long)e*512 + 256 + c];
  #pragma unroll
  for (int dd = 0; dd < 8; dd++){
    float contrib = vec[((long)d*8 + dd)*256 + c]*s1 + dij[(long)e*8 + dd]*s2;
    atomicAdd(vecagg + ((long)s*8 + dd)*256 + c, contrib);
  }
}

__global__ void k_vecdot(const float* v12, float* vd, int Nn){
  long i = (long)blockIdx.x*256 + threadIdx.x; if (i >= (long)Nn*256) return;
  int n = (int)(i >> 8), c = (int)(i & 255);
  float acc = 0.f;
  #pragma unroll
  for (int d = 0; d < 8; d++){
    const float* r = v12 + ((long)n*8 + d)*512;
    acc += r[c]*r[256 + c];
  }
  vd[i] = acc;
}

__global__ void k_updx(float* x, const float* vd, const float* o, int Nn){
  long i = (long)blockIdx.x*256 + threadIdx.x; if (i >= (long)Nn*256) return;
  int n = (int)(i >> 8), c = (int)(i & 255);
  x[i] += vd[i]*o[(long)n*768 + 256 + c] + o[(long)n*768 + 512 + c];
}

__global__ void k_updvec(float* vec, h16* vecH, const float* vec3, const float* vecagg,
                         const float* o, int Nn){
  long i = (long)blockIdx.x*256 + threadIdx.x; if (i >= (long)Nn*8*256) return;
  long r = i >> 8; int c = (int)(i & 255); int n = (int)(r >> 3);
  float nv = vec[i] + vec3[i]*o[(long)n*768 + c] + vecagg[i];
  vec[i] = nv; vecH[i] = (h16)nv;
}

__global__ void k_vecscale(float* vec, h16* vecH, const float* vw, long rows){
  long i = (long)blockIdx.x*256 + threadIdx.x; if (i >= rows*256) return;
  int c = (int)(i & 255);
  float nv = vec[i]*vw[c]; vec[i] = nv; vecH[i] = (h16)nv;
}

__global__ void k_vnorm(const float* v1out, int C, h16* outH, long ldo, long coff, int M){
  long i = (long)blockIdx.x*256 + threadIdx.x; if (i >= (long)M*C) return;
  int m = (int)(i / C), c = (int)(i % C);
  float acc = 1e-16f;
  #pragma unroll
  for (int d = 0; d < 8; d++){ float t = v1out[((long)m*8 + d)*C + c]; acc += t*t; }
  outH[(long)m*ldo + coff + c] = (h16)sqrtf(acc);
}

__global__ void k_gate(const float* u2out, int dout, const float* vec2out,
                       float* cx, float* cvF, h16* cvH, int M, int act){
  long i = (long)blockIdx.x*256 + threadIdx.x; if (i >= (long)M*8*dout) return;
  long r = i / dout; int c = (int)(i % dout); int m = (int)(r >> 3); int d = (int)(r & 7);
  float gate = u2out[(long)m*2*dout + dout + c];
  float cv = gate * vec2out[i];
  if (cvF) cvF[i] = cv;
  if (cvH) cvH[i] = (h16)cv;
  if (d == 0){
    float xo = u2out[(long)m*2*dout + c];
    cx[(long)m*dout + c] = act ? siluf(xo) : xo;
  }
}

__global__ void k_l1v(const float* cv2, const float* w, const float* b, float* l1v, int Nn){
  long i = (long)blockIdx.x*256 + threadIdx.x; if (i >= (long)Nn*3) return;
  int n = (int)(i / 3), k = (int)(i % 3);
  float acc = b[0];
  const float* r = cv2 + ((long)n*8 + k)*64;
  for (int c = 0; c < 64; c++) acc += r[c]*w[c];
  l1v[i] = acc;
}

__global__ void k_l2m(const float* cv2, float* l2m, int Nn){
  long i = (long)blockIdx.x*256 + threadIdx.x; if (i >= (long)Nn*64) return;
  int n = (int)(i >> 6), c = (int)(i & 63);
  float acc = 1e-16f;
  #pragma unroll
  for (int d = 3; d < 8; d++){ float t = cv2[((long)n*8 + d)*64 + c]; acc += t*t; }
  l2m[i] = sqrtf(acc);
}

__global__ void k_coord(const float* t21, const float* l22w, const float* l22b,
                        const float* tsc, const float* sc2w, const float* sc2b,
                        const float* comb, const float* l1v, float* dout, int Nn){
  int n = blockIdx.x*256 + threadIdx.x; if (n >= Nn) return;
  float a = l22b[0]; for (int j = 0; j < 32; j++) a += t21[(long)n*32 + j]*l22w[j];
  float l2mod = tanhf(a);
  float g = sc2b[0]; for (int j = 0; j < 32; j++) g += tsc[(long)n*32 + j]*sc2w[j];
  float mag = sigmf(g);
  float e0 = expf(comb[0]), e1 = expf(comb[1]);
  float w0 = e0/(e0+e1), w1 = e1/(e0+e1);
  float sc = mag*(w0 + w1*l2mod);
  float c0 = sc*l1v[(long)n*3], c1 = sc*l1v[(long)n*3+1], c2 = sc*l1v[(long)n*3+2];
  if (n < NA_C){ dout[(long)n*19]=c0; dout[(long)n*19+1]=c1; dout[(long)n*19+2]=c2; }
  else { long b = (long)NA_C*19 + (long)(n-NA_C)*27; dout[b]=c0; dout[b+1]=c1; dout[b+2]=c2; }
}

__global__ void k_featw(const float* u2, int ldu, float* dout, long obase, int ostride,
                        int ocols, int Mrows){
  long i = (long)blockIdx.x*256 + threadIdx.x; if (i >= (long)Mrows*ocols) return;
  int m = (int)(i / ocols), c = (int)(i % ocols);
  dout[obase + (long)m*ostride + 3 + c] = u2[(long)m*ldu + c];
}

// ---------------------------- host ----------------------------
struct Bump { char* b; size_t o;
  void* get(size_t n){ void* p = b + o; o = (o + n + 255) & ~(size_t)255; return p; } };

extern "C" void kernel_launch(void* const* d_in, const int* in_sizes, int n_in,
                              void* d_out, int out_size, void* d_ws, size_t ws_size,
                              hipStream_t stream){
  (void)in_sizes; (void)n_in; (void)out_size; (void)ws_size;
  const int NN = NN_C, E = E_C, NA = NA_C, NR = NR_C;
  auto PF = [&](int i){ return (const float*)d_in[i]; };
  auto PI = [&](int i){ return (const int*)d_in[i]; };
  // param leaf indices (jax tree_flatten order: dict keys sorted, recursively)
  enum { AE_B1=3,AE_B2,AE_G1,AE_G2,AE_L1B,AE_L1W,AE_L2B,AE_L2W, CV_COMB=11,
         G1_U1B=12,G1_U1W,G1_U2B,G1_U2W,G1_V1W,G1_V2W,
         G2_U1B=18,G2_U1W,G2_U2B,G2_U2W,G2_V1W,G2_V2W,
         CV_L1B=24,CV_L1W,CV_L21B,CV_L21W,CV_L22B,CV_L22W,
         CV_SC1B=30,CV_SC1W,CV_SC2B,CV_SC2W, EE_B=34,EE_W, ETYPE=36,
         FA1_U1B=37,FA1_U1W,FA1_U2B,FA1_U2W,FA1_V1W,FA1_V2W,
         FA2_U1B=43,FA2_U1W,FA2_U2B,FA2_U2W,FA2_V1W,FA2_V2W,
         FR1_U1B=49,FR1_U1W,FR1_U2B,FR1_U2W,FR1_V1W,FR1_V2W,
         FR2_U1B=55,FR2_U1W,FR2_U2B,FR2_U2W,FR2_V1W,FR2_V2W,
         INP_B=61,INP_W, LAYER0=63,
         NBRC_B=189,NBRC_W,NBRD_B,NBRD_W, OUTB=193,OUTG,
         RBF_BETAS=195,RBF_MEANS,
         RE_B1=197,RE_B2,RE_G1,RE_G2,RE_L1B,RE_L1W,RE_L2B,RE_L2W,
         TL_B=205,TL_W, VECW=207, MASK_A=208, MASK_R=209, EIDX=210, ETYP=211, EMASK=212 };
  enum { LDKB=0,LDKW,LDVB,LDVW,LFB,LFW,LKB,LKW,LLNB,LLNG,LOB,LOW,LQB,LQW,LSB,LSW,LVB,LVW,LVECPW,LWSRCW,LWTRGW };
  auto LP = [&](int li, int off){ return (const float*)d_in[LAYER0 + li*21 + off]; };
  const int* ei = PI(EIDX); const int* et = PI(ETYP); const float* em = PF(EMASK);

  Bump al{(char*)d_ws, 0};
  auto prep = [&](const float* W, int ldw, int coff, int K, int Nsub)->h16*{
    h16* WT = (h16*)al.get((size_t)K*Nsub*2);
    long tot = (long)K*Nsub;
    k_prep<<<dim3((unsigned)((tot+255)/256)), dim3(256), 0, stream>>>(W, ldw, coff, K, Nsub, WT);
    return WT;
  };
  auto gemm = [&](const h16* A, long lda, const h16* WT, int K, const float* bias,
                  int M, int Nc, int epi, float* outF, long ldoF, h16* outH, long ldoH,
                  const float* auxC=nullptr, const float* auxX=nullptr, float* auxAcc=nullptr,
                  const float* fijOld=nullptr, const h16* wtrg=nullptr, const h16* wsrc=nullptr){
    GemmP p{A, lda, WT, K, bias, outF, ldoF, outH, ldoH, M, 0, epi,
            ei, E, auxC, auxX, auxAcc, fijOld, em, wtrg, wsrc};
    const unsigned gx = (unsigned)((M/16 + 3)/4);
    const int full = Nc >> 6, rem = (Nc & 63) >> 4;
    if (full){
      p.n0base = 0;
      k_gemm<4><<<dim3(gx, (unsigned)full), dim3(128), 0, stream>>>(p);
    }
    if (rem){
      p.n0base = full * 64;
      dim3 g(gx, 1);
      if (rem == 1)      k_gemm<1><<<g, dim3(128), 0, stream>>>(p);
      else if (rem == 2) k_gemm<2><<<g, dim3(128), 0, stream>>>(p);
      else               k_gemm<3><<<g, dim3(128), 0, stream>>>(p);
    }
  };
  auto g1d = [&](long n){ return dim3((unsigned)((n + 255)/256)); };

  // ---- prep all WMMA weights (f16, transposed) ----
  h16* inpWT  = prep(PF(INP_W), 256, 0, 256, 256);
  h16* nbrdWT = prep(PF(NBRD_W), 256, 0, 32, 256);
  h16* nbrcWT = prep(PF(NBRC_W), 256, 0, 512, 256);
  h16* eeWT   = prep(PF(EE_W), 256, 0, 32, 256);
  h16 *qWT[6], *kWT[6], *vWT[6], *dkWT[6], *dvWT[6], *fWT[6], *oWT[6], *sWT[6],
      *vp12WT[6], *vp3WT[6], *wsWT[6], *wtWT[6];
  for (int li = 0; li < 6; li++){
    qWT[li]  = prep(LP(li,LQW), 256, 0, 256, 256);
    kWT[li]  = prep(LP(li,LKW), 256, 0, 256, 256);
    vWT[li]  = prep(LP(li,LVW), 256, 0, 256, 256);
    dkWT[li] = prep(LP(li,LDKW), 256, 0, 256, 256);
    dvWT[li] = prep(LP(li,LDVW), 256, 0, 256, 256);
    fWT[li]  = prep(LP(li,LFW), 256, 0, 256, 256);
    oWT[li]  = prep(LP(li,LOW), 768, 0, 256, 768);
    sWT[li]  = prep(LP(li,LSW), 512, 0, 256, 512);
    vp12WT[li] = prep(LP(li,LVECPW), 768, 0, 256, 512);
    vp3WT[li]  = prep(LP(li,LVECPW), 768, 512, 256, 256);
    wsWT[li] = prep(LP(li,LWSRCW), 256, 0, 256, 256);
    wtWT[li] = prep(LP(li,LWTRGW), 256, 0, 256, 256);
  }
  h16* g1v1 = prep(PF(G1_V1W), 256, 0, 256, 256);
  h16* g1v2 = prep(PF(G1_V2W), 128, 0, 256, 128);
  h16* g1u1 = prep(PF(G1_U1W), 256, 0, 512, 256);
  h16* g1u2 = prep(PF(G1_U2W), 256, 0, 256, 256);
  h16* g2v1 = prep(PF(G2_V1W), 128, 0, 128, 128);
  h16* g2v2 = prep(PF(G2_V2W), 64, 0, 128, 64);
  h16* g2u1 = prep(PF(G2_U1W), 128, 0, 256, 128);
  h16* g2u2 = prep(PF(G2_U2W), 128, 0, 128, 128);
  h16* fa1v1 = prep(PF(FA1_V1W), 256, 0, 256, 256);
  h16* fa1v2 = prep(PF(FA1_V2W), 128, 0, 256, 128);
  h16* fa1u1 = prep(PF(FA1_U1W), 256, 0, 512, 256);
  h16* fa1u2 = prep(PF(FA1_U2W), 256, 0, 256, 256);
  h16* fa2v1 = prep(PF(FA2_V1W), 128, 0, 128, 128);
  h16* fa2u1 = prep(PF(FA2_U1W), 128, 0, 256, 128);
  h16* fa2u2 = prep(PF(FA2_U2W), 32, 0, 128, 32);
  h16* fr1v1 = prep(PF(FR1_V1W), 256, 0, 256, 256);
  h16* fr1v2 = prep(PF(FR1_V2W), 128, 0, 256, 128);
  h16* fr1u1 = prep(PF(FR1_U1W), 256, 0, 512, 256);
  h16* fr1u2 = prep(PF(FR1_U2W), 256, 0, 256, 256);
  h16* fr2v1 = prep(PF(FR2_V1W), 128, 0, 128, 128);
  h16* fr2u1 = prep(PF(FR2_U1W), 128, 0, 256, 128);
  h16* fr2u2 = prep(PF(FR2_U2W), 48, 0, 128, 48);

  // ---- buffers ----
  float* x    = (float*)al.get((size_t)NN*256*4);
  float* vec  = (float*)al.get((size_t)NN*8*256*4);
  h16*  vecH  = (h16*)al.get((size_t)NN*8*256*2);
  float* fij  = (float*)al.get((size_t)E*256*4);
  h16*  fijHA = (h16*)al.get((size_t)E*256*2);
  h16*  fijHB = (h16*)al.get((size_t)E*256*2);
  float* dij  = (float*)al.get((size_t)E*8*4);
  float* Cbuf = (float*)al.get((size_t)E*4);
  float* teval= (float*)al.get(256);
  float* bias2= (float*)al.get(256*4);
  float* hbuf = (float*)al.get((size_t)NN*256*4);
  h16*  hH    = (h16*)al.get((size_t)NN*256*2);
  h16*  eattrH= (h16*)al.get((size_t)E*32*2);
  float* qb = (float*)al.get((size_t)NN*256*4);
  float* kb = (float*)al.get((size_t)NN*256*4);
  float* vb = (float*)al.get((size_t)NN*256*4);
  float* vdot = (float*)al.get((size_t)NN*256*4);
  float* attn = (float*)al.get((size_t)E*8*4);
  float* obuf = (float*)al.get((size_t)NN*768*4);
  float* xagg = (float*)al.get((size_t)NN*256*4);
  h16*  xaggH = (h16*)al.get((size_t)NN*256*2);
  float* vagg = (float*)al.get((size_t)NN*8*256*4);
  float* vec3 = (float*)al.get((size_t)NN*8*256*4);
  float* arenaA = (float*)al.get((size_t)NN*8*512*4);     // 32 MB, reused
  float* arenaB = (float*)al.get((size_t)E*256*2 + 4096); // 12.6 MB, reused
  h16* concatH = (h16*)al.get((size_t)NN*512*2);
  float* encT = (float*)al.get((size_t)NR*128*4);
  float* l1vb = (float*)al.get((size_t)NN*3*4);
  float* l2mb = (float*)al.get((size_t)NN*64*4);
  float* t21 = (float*)al.get((size_t)NN*32*4);
  float* tsc = (float*)al.get((size_t)NN*32*4);

  // ---- phase 0: encoders, time embedding, input projection ----
  k_te<<<dim3(1), dim3(128), 0, stream>>>(PF(2), PF(TL_W), PF(TL_B), teval);
  k_bias2<<<dim3(1), dim3(256), 0, stream>>>(PF(INP_B), PF(INP_W), teval, bias2);
  // atoms: l1 -> ln+silu -> l2 -> ln (f32 + f16)
  k_nlin<<<g1d((long)NA*128), dim3(256), 0, stream>>>(PF(0), 19, 3, 16, 128, PF(AE_L1W), PF(AE_L1B), encT, NA, 0);
  k_lnorm<<<dim3(NA), dim3(128), 0, stream>>>(encT, PF(AE_G1), PF(AE_B1), encT, nullptr, 128, 1);
  k_nlin<<<g1d((long)NA*256), dim3(256), 0, stream>>>(encT, 128, 0, 128, 256, PF(AE_L2W), PF(AE_L2B), hbuf, NA, 0);
  k_lnorm<<<dim3(NA), dim3(256), 0, stream>>>(hbuf, PF(AE_G2), PF(AE_B2), hbuf, hH, 256, 0);
  // residues
  k_nlin<<<g1d((long)NR*128), dim3(256), 0, stream>>>(PF(1), 27, 3, 24, 128, PF(RE_L1W), PF(RE_L1B), encT, NR, 0);
  k_lnorm<<<dim3(NR), dim3(128), 0, stream>>>(encT, PF(RE_G1), PF(RE_B1), encT, nullptr, 128, 1);
  k_nlin<<<g1d((long)NR*256), dim3(256), 0, stream>>>(encT, 128, 0, 128, 256, PF(RE_L2W), PF(RE_L2B), hbuf + (size_t)NA*256, NR, 0);
  k_lnorm<<<dim3(NR), dim3(256), 0, stream>>>(hbuf + (size_t)NA*256, PF(RE_G2), PF(RE_B2),
      hbuf + (size_t)NA*256, hH + (size_t)NA*256, 256, 0);
  // x = h @ inp.W[:256] + (b + te*inp.W[256])
  gemm(hH, 256, inpWT, 256, bias2, NN, 256, EPI_PLAIN, x, 256, nullptr, 0);
  // edge geometry
  k_edge<<<g1d(E), dim3(256), 0, stream>>>(ei, et, em, PF(0), PF(1), PF(RBF_BETAS), PF(RBF_MEANS),
                                           PF(ETYPE), Cbuf, dij, eattrH, E);
  // msg = segsum(x[dst]*Wd, src), Wd fused into nbr_dist GEMM epilogue
  hipMemsetAsync(xagg, 0, (size_t)NN*256*4, stream);
  gemm(eattrH, 32, nbrdWT, 32, PF(NBRD_B), E, 256, EPI_NBR, nullptr, 0, nullptr, 0,
       Cbuf, x, xagg);
  // x = nbr_comb(concat[x, msg])
  k_cvt<<<g1d((long)NN*256), dim3(256), 0, stream>>>(x, 256, concatH, 512, 0, NN, 256);
  k_cvt<<<g1d((long)NN*256), dim3(256), 0, stream>>>(xagg, 256, concatH, 512, 256, NN, 256);
  gemm(concatH, 512, nbrcWT, 512, PF(NBRC_B), NN, 256, EPI_PLAIN, x, 256, nullptr, 0);
  // f_ij = edge_emb(edge_attr) * (x[src]+x[dst])
  gemm(eattrH, 32, eeWT, 32, PF(EE_B), E, 256, EPI_EDGEEMB, fij, 256, fijHA, 256,
       nullptr, x);
  hipMemsetAsync(vec, 0, (size_t)NN*8*256*4, stream);
  hipMemsetAsync(vecH, 0, (size_t)NN*8*256*2, stream);

  // ---- layers ----
  h16* fc = fijHA; h16* fn = fijHB;
  for (int li = 0; li < 6; li++){
    k_lnorm<<<dim3(NN), dim3(256), 0, stream>>>(x, LP(li,LLNG), LP(li,LLNB), nullptr, hH, 256, 0);
    gemm(hH, 256, qWT[li], 256, LP(li,LQB), NN, 256, EPI_PLAIN, qb, 256, nullptr, 0);
    gemm(hH, 256, kWT[li], 256, LP(li,LKB), NN, 256, EPI_PLAIN, kb, 256, nullptr, 0);
    gemm(hH, 256, vWT[li], 256, LP(li,LVB), NN, 256, EPI_PLAIN, vb, 256, nullptr, 0);
    // vec1|vec2, vec_dot, vec3
    gemm(vecH, 256, vp12WT[li], 256, nullptr, NN*8, 512, EPI_PLAIN, arenaA, 512, nullptr, 0);
    k_vecdot<<<g1d((long)NN*256), dim3(256), 0, stream>>>(arenaA, vdot, NN);
    gemm(vecH, 256, vp3WT[li], 256, nullptr, NN*8, 256, EPI_PLAIN, vec3, 256, nullptr, 0);
    // dk, dv (f16, silu)
    h16* dkH = (h16*)arenaA; h16* dvH = dkH + (size_t)E*256;
    gemm(fc, 256, dkWT[li], 256, LP(li,LDKB), E, 256, EPI_SILU_F16, nullptr, 0, dkH, 256);
    gemm(fc, 256, dvWT[li], 256, LP(li,LDVB), E, 256, EPI_SILU_F16, nullptr, 0, dvH, 256);
    k_attn<<<g1d((long)E*8), dim3(256), 0, stream>>>(qb, kb, dkH, Cbuf, ei, attn, E);
    hipMemsetAsync(xagg, 0, (size_t)NN*256*4, stream);
    h16* vmsgH = (h16*)arenaB;
    k_vmsg<<<g1d((long)E*256), dim3(256), 0, stream>>>(vb, dvH, attn, ei, vmsgH, xagg, E);
    // s1|s2 (f16, silu)
    h16* sH = (h16*)arenaA;
    gemm(vmsgH, 256, sWT[li], 256, LP(li,LSB), E, 512, EPI_SILU_F16, nullptr, 0, sH, 512);
    hipMemsetAsync(vagg, 0, (size_t)NN*8*256*4, stream);
    k_vscat<<<g1d((long)E*256), dim3(256), 0, stream>>>(vec, sH, dij, em, ei, vagg, E);
    // o(x_agg)
    k_cvt<<<g1d((long)NN*256), dim3(256), 0, stream>>>(xagg, 256, xaggH, 256, 0, NN, 256);
    gemm(xaggH, 256, oWT[li], 256, LP(li,LOB), NN, 768, EPI_PLAIN, obuf, 768, nullptr, 0);
    if (li < 5){
      h16* wsrcH = (h16*)arenaA; h16* wtrgH = wsrcH + (size_t)NN*8*256;
      gemm(vecH, 256, wsWT[li], 256, nullptr, NN*8, 256, EPI_F16, nullptr, 0, wsrcH, 256);
      gemm(vecH, 256, wtWT[li], 256, nullptr, NN*8, 256, EPI_F16, nullptr, 0, wtrgH, 256);
      gemm(fc, 256, fWT[li], 256, LP(li,LFB), E, 256, EPI_FUPD, fij, 256, fn, 256,
           nullptr, nullptr, nullptr, fij, wtrgH, wsrcH);
      h16* tmp = fc; fc = fn; fn = tmp;
    }
    k_updx<<<g1d((long)NN*256), dim3(256), 0, stream>>>(x, vdot, obuf, NN);
    k_updvec<<<g1d((long)NN*8*256), dim3(256), 0, stream>>>(vec, vecH, vec3, vagg, obuf, NN);
  }

  // ---- output head ----
  k_lnorm<<<dim3(NN), dim3(256), 0, stream>>>(x, PF(OUTG), PF(OUTB), x, nullptr, 256, 0);
  k_vecscale<<<g1d((long)NN*8*256), dim3(256), 0, stream>>>(vec, vecH, PF(VECW), (long)NN*8);
  // cv_geb1
  gemm(vecH, 256, g1v1, 256, nullptr, NN*8, 256, EPI_PLAIN, arenaA, 256, nullptr, 0);
  k_vnorm<<<g1d((long)NN*256), dim3(256), 0, stream>>>(arenaA, 256, concatH, 512, 256, NN);
  k_cvt<<<g1d((long)NN*256), dim3(256), 0, stream>>>(x, 256, concatH, 512, 0, NN, 256);
  gemm(vecH, 256, g1v2, 256, nullptr, NN*8, 128, EPI_PLAIN, arenaB, 128, nullptr, 0);
  gemm(concatH, 512, g1u1, 512, PF(G1_U1B), NN, 256, EPI_SILU_F16, nullptr, 0, hH, 256);
  gemm(hH, 256, g1u2, 256, PF(G1_U2B), NN, 256, EPI_PLAIN, qb, 256, nullptr, 0);
  k_gate<<<g1d((long)NN*8*128), dim3(256), 0, stream>>>(qb, 128, arenaB, kb, nullptr, (h16*)vec3, NN, 1);
  // cv_geb2
  gemm((h16*)vec3, 128, g2v1, 128, nullptr, NN*8, 128, EPI_PLAIN, arenaA, 128, nullptr, 0);
  k_vnorm<<<g1d((long)NN*128), dim3(256), 0, stream>>>(arenaA, 128, concatH, 256, 128, NN);
  k_cvt<<<g1d((long)NN*128), dim3(256), 0, stream>>>(kb, 128, concatH, 256, 0, NN, 128);
  gemm((h16*)vec3, 128, g2v2, 128, nullptr, NN*8, 64, EPI_PLAIN, arenaB, 64, nullptr, 0);
  gemm(concatH, 256, g2u1, 256, PF(G2_U1B), NN, 128, EPI_SILU_F16, nullptr, 0, hH, 128);
  gemm(hH, 128, g2u2, 128, PF(G2_U2B), NN, 128, EPI_PLAIN, qb, 128, nullptr, 0);
  k_gate<<<g1d((long)NN*8*64), dim3(256), 0, stream>>>(qb, 64, arenaB, vb, vagg, nullptr, NN, 1);
  // coord
  k_l1v<<<g1d((long)NN*3), dim3(256), 0, stream>>>(vagg, PF(CV_L1W), PF(CV_L1B), l1vb, NN);
  k_l2m<<<g1d((long)NN*64), dim3(256), 0, stream>>>(vagg, l2mb, NN);
  k_nlin<<<g1d((long)NN*32), dim3(256), 0, stream>>>(l2mb, 64, 0, 64, 32, PF(CV_L21W), PF(CV_L21B), t21, NN, 1);
  k_nlin<<<g1d((long)NN*32), dim3(256), 0, stream>>>(vb, 64, 0, 64, 32, PF(CV_SC1W), PF(CV_SC1B), tsc, NN, 1);
  k_coord<<<g1d(NN), dim3(256), 0, stream>>>(t21, PF(CV_L22W), PF(CV_L22B), tsc, PF(CV_SC2W),
      PF(CV_SC2B), PF(CV_COMB), l1vb, (float*)d_out, NN);

  // ---- feature heads (fa: atoms, fr: residues) ----
  struct FeatW { const h16 *v1a, *v2a, *u1a, *u2a, *v1b, *u1b, *u2b;
                 int U1Ba, U2Ba, U1Bb, U2Bb; };
  FeatW FA{fa1v1, fa1v2, fa1u1, fa1u2, fa2v1, fa2u1, fa2u2, FA1_U1B, FA1_U2B, FA2_U1B, FA2_U2B};
  FeatW FR{fr1v1, fr1v2, fr1u1, fr1u2, fr2v1, fr2u1, fr2u2, FR1_U1B, FR1_U2B, FR2_U1B, FR2_U2B};
  for (int half = 0; half < 2; half++){
    const FeatW& W = half ? FR : FA;
    const int M = half ? NR : NA;
    const int rowoff = half ? NA : 0;
    const int u2n = half ? 48 : 32;
    const int ocols = half ? 24 : 16;
    const int ostride = half ? 27 : 19;
    const long obase = half ? (long)NA*19 : 0;
    const h16* vH = vecH + (size_t)rowoff*8*256;
    // geb1 (din 256, dout 128, scalar_act)
    gemm(vH, 256, W.v1a, 256, nullptr, M*8, 256, EPI_PLAIN, arenaA, 256, nullptr, 0);
    k_vnorm<<<g1d((long)M*256), dim3(256), 0, stream>>>(arenaA, 256, concatH, 512, 256, M);
    k_cvt<<<g1d((long)M*256), dim3(256), 0, stream>>>(x + (size_t)rowoff*256, 256, concatH, 512, 0, M, 256);
    gemm(vH, 256, W.v2a, 256, nullptr, M*8, 128, EPI_PLAIN, arenaB, 128, nullptr, 0);
    gemm(concatH, 512, W.u1a, 512, PF(W.U1Ba), M, 256, EPI_SILU_F16, nullptr, 0, hH, 256);
    gemm(hH, 256, W.u2a, 256, PF(W.U2Ba), M, 256, EPI_PLAIN, qb, 256, nullptr, 0);
    k_gate<<<g1d((long)M*8*128), dim3(256), 0, stream>>>(qb, 128, arenaB, kb, nullptr, (h16*)vec3, M, 1);
    // geb2 (din 128, no scalar act, only xo needed -> skip v2/vo)
    gemm((h16*)vec3, 128, W.v1b, 128, nullptr, M*8, 128, EPI_PLAIN, arenaA, 128, nullptr, 0);
    k_vnorm<<<g1d((long)M*128), dim3(256), 0, stream>>>(arenaA, 128, concatH, 256, 128, M);
    k_cvt<<<g1d((long)M*128), dim3(256), 0, stream>>>(kb, 128, concatH, 256, 0, M, 128);
    gemm(concatH, 256, W.u1b, 256, PF(W.U1Bb), M, 128, EPI_SILU_F16, nullptr, 0, hH, 128);
    gemm(hH, 128, W.u2b, 128, PF(W.U2Bb), M, u2n, EPI_PLAIN, qb, u2n, nullptr, 0);
    k_featw<<<g1d((long)M*ocols), dim3(256), 0, stream>>>(qb, u2n, (float*)d_out, obase, ostride, ocols, M);
  }
}
